// MoEGate_85332410237528
// MI455X (gfx1250) — compile-verified, matrix operand order checked by reading, not run on hardware
//
#include <hip/hip_runtime.h>
#include <hip/hip_bf16.h>
#include <stdint.h>

// ---------------- problem constants ----------------
#define DIM     7168
#define NEXP    256
#define TOPK    8
#define NTOK    8192
#define RSCALE  2.5f

#define MBLK    32                 // tokens per workgroup
#define KSTEP   32                 // K per pipeline stage (== WMMA bf16 K)
#define NSTEPS  (DIM / KSTEP)      // 224
#define NWG     (NTOK / MBLK)      // 256 workgroups

// padded LDS row: 32 bf16 payload + 8 bf16 pad = 20 dwords.
// 20 dwords: 16B-aligned rows (20%4==0) and gcd(20,64)==4 -> the 16 distinct
// row-start banks of a fragment load cover all 64 banks: conflict-free b128.
#define ROW_DW  20
#define A_DW    (MBLK * ROW_DW)    // 640 dwords
#define B_DW    (NEXP * ROW_DW)    // 5120 dwords
#define BUF_DW  (A_DW + B_DW)      // 5760 dwords per buffer
#define LDS_DW  (2 * BUF_DW)       // 11520 dwords = 46 KB

#define LOG_STRIDE 257             // padded logits row (odd -> conflict-free columns)

typedef __attribute__((ext_vector_type(16))) __bf16 v16bf;
typedef __attribute__((ext_vector_type(2)))  __bf16 v2bf;
typedef __attribute__((ext_vector_type(8)))  float  v8f;

union FragBF { uint4 q[2]; v16bf v; };

#if __has_builtin(__builtin_amdgcn_global_load_async_to_lds_b128)
#define HAVE_ASYNC 1
#else
#define HAVE_ASYNC 0
#endif

// f32 -> bf16 (RNE via native fptrunc -> v_cvt_pk_bf16_f32), packed into one dword
__device__ inline unsigned f2bf_pk(float a, float b) {
  v2bf v;
  v[0] = (__bf16)a;
  v[1] = (__bf16)b;
  unsigned u;
  __builtin_memcpy(&u, &v, 4);
  return u;
}

#if HAVE_ASYNC
// builtin signature (from hipcc diagnostic): param0 = v4i AS1*, param1 = v4i AS3*
typedef int v4i __attribute__((vector_size(16)));
typedef v4i __attribute__((address_space(1)))* gbl_v4i;
typedef v4i __attribute__((address_space(3)))* lds_v4i;

__device__ inline void asyncCopy16(const void* g, void* l) {
  __builtin_amdgcn_global_load_async_to_lds_b128((gbl_v4i)g, (lds_v4i)l, 0, 0);
}
#endif

__device__ inline void waitAsync() {
#if __has_builtin(__builtin_amdgcn_s_wait_asynccnt)
  __builtin_amdgcn_s_wait_asynccnt(0);
#else
  asm volatile("s_wait_asynccnt 0x0" ::: "memory");
#endif
}

// ---------------- kernel 1: gate_w f32 -> bf16 (once) ----------------
__global__ __launch_bounds__(256) void gw_to_bf16(const float4* __restrict__ src,
                                                  uint2* __restrict__ dst, int n4) {
  int i = blockIdx.x * 256 + threadIdx.x;
  if (i < n4) {
    float4 v = src[i];
    uint2 r;
    r.x = f2bf_pk(v.x, v.y);
    r.y = f2bf_pk(v.z, v.w);
    dst[i] = r;
  }
}

// 16-bit WMMA fragment from a padded LDS row (ISA 7.12.2 layout):
// lane l: row (l&15), kbase=(l>>4)*8 elems -> two 16B chunks at dword kbase/2 and 8+kbase/2
__device__ inline v16bf loadFrag(const unsigned* __restrict__ rowbase, int lane) {
  FragBF f;
  int kb = (lane >> 4) * 4;                            // dwords
  f.q[0] = *(const uint4*)(rowbase + kb);
  f.q[1] = *(const uint4*)(rowbase + 8 + kb);
  return f.v;
}

// ---------------- kernel 2: fused GEMM + softmax + top-k ----------------
__global__ __launch_bounds__(256) void moe_gate_gemm(
    const float* __restrict__ x, const unsigned* __restrict__ gwbf,
    float* __restrict__ outW, int* __restrict__ outI,
    float* __restrict__ fPart, float* __restrict__ pPart) {
  __shared__ __align__(16) unsigned lds[LDS_DW];

  const int tid  = threadIdx.x;
  const int lane = tid & 31;
  const int wid  = tid >> 5;
  const int wg   = blockIdx.x;
  const int tok0 = wg * MBLK;

  // staging map: A row am (0..31), chunk ac (0..7) ; B row e = tid (0..255)
  const int am = tid >> 3, ac = tid & 7;
  const float4* xrow4 = (const float4*)(x + (size_t)(tok0 + am) * DIM);
  const unsigned char* gwrowB =
      (const unsigned char*)gwbf + (size_t)tid * (DIM * 2);   // bf16 row, bytes
#if !HAVE_ASYNC
  const uint4* gwrow4 = (const uint4*)gwrowB;
#endif

  // tile map: 2 M-tiles x 16 N-tiles over 8 waves -> 4 tiles per wave
  const int mtile = wid & 1;
  const int n0    = (wid >> 1) * 4;

  v8f acc0 = {}, acc1 = {}, acc2 = {}, acc3 = {};

  // ---- prologue: stage tile 0 into buf0 ----
  {
    unsigned* bp = lds + A_DW + tid * ROW_DW;
#if HAVE_ASYNC
#pragma unroll
    for (int j = 0; j < 4; ++j) asyncCopy16(gwrowB + j * 16, bp + j * 4);
#else
#pragma unroll
    for (int j = 0; j < 4; ++j) *(uint4*)(bp + j * 4) = gwrow4[j];
#endif
    float4 a4 = xrow4[ac];
    unsigned* ap = lds + am * ROW_DW + ac * 2;
    ap[0] = f2bf_pk(a4.x, a4.y);
    ap[1] = f2bf_pk(a4.z, a4.w);
#if HAVE_ASYNC
    waitAsync();
#endif
  }
  float4 a4n = xrow4[8 + ac];                  // A for tile 1, distance-2 prefetch
  __syncthreads();

#pragma unroll 1
  for (int kt = 0; kt < NSTEPS; ++kt) {
    unsigned* cur = lds + (kt & 1) * BUF_DW;
    unsigned* nxt = lds + ((kt + 1) & 1) * BUF_DW;
    const bool more = (kt + 1) < NSTEPS;

#if HAVE_ASYNC
    if (more) {
      const unsigned char* src = gwrowB + (size_t)(kt + 1) * 64;
      unsigned* bp = nxt + A_DW + tid * ROW_DW;
#pragma unroll
      for (int j = 0; j < 4; ++j) asyncCopy16(src + j * 16, bp + j * 4);
    }
#else
    uint4 b[4];
    if (more) {
      const uint4* p = gwrow4 + (kt + 1) * 4;
      b[0] = p[0]; b[1] = p[1]; b[2] = p[2]; b[3] = p[3];
    }
#endif
    // prefetch A for tile kt+2 (clamped); consumed one iteration later, so its
    // latency is hidden under a full compute + barrier interval
    const int kpre = (kt + 2 < NSTEPS) ? (kt + 2) : (NSTEPS - 1);
    float4 a4nn = xrow4[kpre * 8 + ac];
    __builtin_amdgcn_sched_barrier(0);         // keep staging issues above compute

    // ---- compute: fragments + back-to-back WMMAs ----
    const int rsel = lane & 15;
    v16bf afr = loadFrag(cur + (mtile * 16 + rsel) * ROW_DW, lane);
    v16bf bf0 = loadFrag(cur + A_DW + ((n0 + 0) * 16 + rsel) * ROW_DW, lane);
    v16bf bf1 = loadFrag(cur + A_DW + ((n0 + 1) * 16 + rsel) * ROW_DW, lane);
    v16bf bf2 = loadFrag(cur + A_DW + ((n0 + 2) * 16 + rsel) * ROW_DW, lane);
    v16bf bf3 = loadFrag(cur + A_DW + ((n0 + 3) * 16 + rsel) * ROW_DW, lane);
    acc0 = __builtin_amdgcn_wmma_f32_16x16x32_bf16(false, afr, false, bf0, (short)0, acc0, false, false);
    acc1 = __builtin_amdgcn_wmma_f32_16x16x32_bf16(false, afr, false, bf1, (short)0, acc1, false, false);
    acc2 = __builtin_amdgcn_wmma_f32_16x16x32_bf16(false, afr, false, bf2, (short)0, acc2, false, false);
    acc3 = __builtin_amdgcn_wmma_f32_16x16x32_bf16(false, afr, false, bf3, (short)0, acc3, false, false);
    __builtin_amdgcn_sched_barrier(0);

    if (more) {
      unsigned* ap = nxt + am * ROW_DW + ac * 2;       // store prefetched A
      ap[0] = f2bf_pk(a4n.x, a4n.y);
      ap[1] = f2bf_pk(a4n.z, a4n.w);
#if !HAVE_ASYNC
      unsigned* bp = nxt + A_DW + tid * ROW_DW;
      *(uint4*)(bp + 0)  = b[0];
      *(uint4*)(bp + 4)  = b[1];
      *(uint4*)(bp + 8)  = b[2];
      *(uint4*)(bp + 12) = b[3];
#endif
    }
#if HAVE_ASYNC
    if (more) waitAsync();
#endif
    __syncthreads();
    a4n = a4nn;
  }

  // ---- epilogue: staging LDS is dead; reuse for logits [32][257] + counts ----
  float* Ls   = (float*)lds;
  float* fcnt = (float*)lds + MBLK * LOG_STRIDE;       // 8224..8479 < 11520

  {
    v8f accs[4] = {acc0, acc1, acc2, acc3};
#pragma unroll
    for (int j = 0; j < 4; ++j) {
      int n = (n0 + j) * 16 + (lane & 15);
#pragma unroll
      for (int r = 0; r < 8; ++r) {
        int m = mtile * 16 + r + ((lane >> 4) * 8);    // C layout (ISA 7.12.2)
        Ls[m * LOG_STRIDE + n] = RSCALE * accs[j][r];
      }
    }
  }
  fcnt[tid] = 0.0f;
  __syncthreads();

  if (tid < MBLK) {
    float* row = Ls + tid * LOG_STRIDE;
    float mx = -1e30f;
    for (int e = 0; e < NEXP; ++e) mx = fmaxf(mx, row[e]);

    float bv[TOPK]; int bi[TOPK];
#pragma unroll
    for (int j = 0; j < TOPK; ++j) { bv[j] = -1e30f; bi[j] = -1; }
    float sum = 0.f;
    for (int e = 0; e < NEXP; ++e) {
      float l = row[e];
      sum += __expf(l - mx);
      if (l > bv[TOPK - 1]) {                          // strict '>' => stable ties
        int p = TOPK - 1;
        while (p > 0 && l > bv[p - 1]) { bv[p] = bv[p - 1]; bi[p] = bi[p - 1]; --p; }
        bv[p] = l; bi[p] = e;
      }
    }
    float inv = 1.0f / sum;
    for (int e = 0; e < NEXP; ++e) row[e] = __expf(row[e] - mx) * inv;  // scores

    float wv[TOPK]; float wsum = 0.f;
#pragma unroll
    for (int j = 0; j < TOPK; ++j) { wv[j] = __expf(bv[j] - mx) * inv; wsum += wv[j]; }
    float winv = 1.0f / wsum;
    int gt = tok0 + tid;
#pragma unroll
    for (int j = 0; j < TOPK; ++j) {
      outW[gt * TOPK + j] = wv[j] * winv;
      outI[gt * TOPK + j] = bi[j];
      atomicAdd(&fcnt[bi[j]], 1.0f);                   // exact integer adds
    }
  }
  __syncthreads();

  {   // deterministic per-WG, per-expert partials
    float ps = 0.f;
    for (int t = 0; t < MBLK; ++t) ps += Ls[t * LOG_STRIDE + tid];
    pPart[wg * NEXP + tid] = ps;
    fPart[wg * NEXP + tid] = fcnt[tid];
  }
}

// ---------------- kernel 3: aux loss reduction ----------------
__global__ __launch_bounds__(256) void moe_gate_aux(const float* __restrict__ fPart,
                                                    const float* __restrict__ pPart,
                                                    float* __restrict__ auxOut) {
  __shared__ float red[256];
  int e = threadIdx.x;
  float F = 0.f, P = 0.f;
  for (int w = 0; w < NWG; ++w) { F += fPart[w * NEXP + e]; P += pPart[w * NEXP + e]; }
  red[e] = F * P;
  __syncthreads();
  for (int s = 128; s > 0; s >>= 1) {
    if (e < s) red[e] += red[e + s];
    __syncthreads();
  }
  if (e == 0) *auxOut = red[0] * ((float)NEXP / ((float)NTOK * (float)NTOK));
}

// ---------------- launch ----------------
extern "C" void kernel_launch(void* const* d_in, const int* in_sizes, int n_in,
                              void* d_out, int out_size, void* d_ws, size_t ws_size,
                              hipStream_t stream) {
  const float* x  = (const float*)d_in[0];
  const float* gw = (const float*)d_in[1];

  // outputs concatenated flat: topk_w (f32, 65536), topk_idx (i32, 65536), aux (f32, 1)
  float* outW = (float*)d_out;
  int*   outI = (int*)((float*)d_out + NTOK * TOPK);
  float* aux  = (float*)d_out + 2 * NTOK * TOPK;

  unsigned char* ws = (unsigned char*)d_ws;
  unsigned* gwbf = (unsigned*)ws;                              // 3,670,016 B (bf16 gate_w)
  float* fPart   = (float*)(ws + 3670016);                     // 256 KB
  float* pPart   = (float*)(ws + 3670016 + 262144);            // 256 KB

  int n4 = NEXP * DIM / 4;                                     // 458752 float4s
  gw_to_bf16<<<(n4 + 255) / 256, 256, 0, stream>>>((const float4*)gw, (uint2*)gwbf, n4);
  moe_gate_gemm<<<NWG, 256, 0, stream>>>(x, gwbf, outW, outI, fPart, pPart);
  moe_gate_aux<<<1, 256, 0, stream>>>(fPart, pPart, aux);
}
